// EvoformerIteration_2199023255618
// MI455X (gfx1250) — matrix operand order repaired and checked by reading, not compiled, not run on hardware
//
#include <hip/hip_runtime.h>
#include <hip/hip_bf16.h>

// =====================================================================
// Evoformer iteration for MI455X (gfx1250).
// All matmuls run through v_wmma_f32_16x16x32_bf16 (wave32, 16x16 tiles).
// Compute-bound problem (~65 GFLOP over ~150MB) -> matrix pipe in bf16,
// f32 accumulation. 192MB L2 caches the whole working set; fragments are
// fetched as 16B global_load_b128 in the ISA-documented WMMA layouts.
// Each wave register-tiles a 32x32 output macro-tile: 4 WMMAs per K-step
// against 8 b128 loads. Full-tile GEMMs take a branch-free fast-path
// K-loop; edge tiles use a scalar-guarded loop (readfirstlane keeps all
// guards in SGPRs so EXEC stays all-1s around WMMA per ISA 7.12).
// Workspace requirement: ~430 MB (bump-allocated, deterministic).
// =====================================================================

#define ACT_RELU 1
#define F_ACCUM  2
#define FAC 0.17677669529663687f   // 1/sqrt(AC), AC=32

typedef __attribute__((ext_vector_type(16))) __bf16 v16bf;
typedef __attribute__((ext_vector_type(8)))  float  v8f;

union FragBF { v16bf v; uint4 q[2]; };
union FragF  { v8f  v; float f[8]; };

__device__ __forceinline__ unsigned short f2bf(float f) {
  unsigned u = __builtin_bit_cast(unsigned, f);
  u += 0x7FFFu + ((u >> 16) & 1u);          // round-to-nearest-even
  return (unsigned short)(u >> 16);
}

// ----------------------------------------------------------------------
// Generic batched WMMA GEMM:
//   C[row*ldc + col*cstride] (+)= scale * (A[M,K] @ Bt[N,K]^T) + bias[col]
// A,Bt bf16 (stored as ushort), C f32. 128 threads = 4 waves; each wave
// owns a 32x32 macro-tile (2x2 accumulators sharing A/B fragments), the
// block covers 64x64. K multiple of 32; M,N multiples of 16 (N=8 allowed
// via zero-padded Bt rows + guarded stores). batch via blockIdx.z strides.
// Fragment layouts per CDNA5 ISA 7.12.2:
//   A (16x32 bf16): lanes 0-15 row M=l, K0-7 (v0-3) + K16-23 (v4-7);
//                   lanes 16-31 row M=l-16, K8-15 + K24-31.
//   B (32x16 bf16): lanes 0-15 col N=l, K0-7 + K16-23; lanes 16-31 col, K8-15 + K24-31.
// ----------------------------------------------------------------------
__global__ __launch_bounds__(128)
void k_gemm(const unsigned short* __restrict__ A, const unsigned short* __restrict__ Bt,
            float* __restrict__ C, int M, int N, int K, int lda, int ldb,
            long long ldc, int cstride, const float* __restrict__ bias,
            float scale, int flags, long long sA, long long sB, long long sC)
{
  int z = blockIdx.z;
  A  += (long long)z * sA;
  Bt += (long long)z * sB;
  C  += (long long)z * sC;
  // readfirstlane -> wave id lives in an SGPR, so every tile guard below is
  // a scalar branch (s_cmp/s_cbranch) and EXEC is never masked around WMMA.
  int wave = __builtin_amdgcn_readfirstlane((int)threadIdx.x) >> 5;
  int lane = threadIdx.x & 31;
  int tileM = blockIdx.y * 64 + (wave >> 1) * 32;
  int tileN = blockIdx.x * 64 + (wave & 1) * 32;
  if (tileM >= M || tileN >= N) return;     // scalar
  bool m1ok = (tileM + 16) < M;             // scalar
  bool n1ok = (tileN + 16) < N;             // scalar
  int l15 = lane & 15, hi = lane >> 4;
  const unsigned short* ap0 = A  + (long long)(tileM + l15) * lda + hi * 8;
  const unsigned short* ap1 = ap0 + (long long)16 * lda;
  const unsigned short* bp0 = Bt + (long long)(tileN + l15) * ldb + hi * 8;
  const unsigned short* bp1 = bp0 + (long long)16 * ldb;
  FragF a00, a01, a10, a11;
#pragma unroll
  for (int p = 0; p < 8; ++p) { a00.f[p] = 0.f; a01.f[p] = 0.f; a10.f[p] = 0.f; a11.f[p] = 0.f; }

  if (m1ok && n1ok) {
    // -------- fast path: full 32x32 macro-tile, branch-free K-loop --------
    for (int k0 = 0; k0 < K; k0 += 32) {
      FragBF fa0, fa1, fb0, fb1;
      fa0.q[0] = *(const uint4*)(ap0);
      fa0.q[1] = *(const uint4*)(ap0 + 16);
      fa1.q[0] = *(const uint4*)(ap1);
      fa1.q[1] = *(const uint4*)(ap1 + 16);
      fb0.q[0] = *(const uint4*)(bp0);
      fb0.q[1] = *(const uint4*)(bp0 + 16);
      fb1.q[0] = *(const uint4*)(bp1);
      fb1.q[1] = *(const uint4*)(bp1 + 16);
      ap0 += 32; ap1 += 32; bp0 += 32; bp1 += 32;
      if (k0 + 32 < K) { __builtin_prefetch(ap0, 0, 1); __builtin_prefetch(bp0, 0, 1); }
      a00.v = __builtin_amdgcn_wmma_f32_16x16x32_bf16(false, fa0.v, false, fb0.v,
                                                      (short)0, a00.v, false, false);
      a01.v = __builtin_amdgcn_wmma_f32_16x16x32_bf16(false, fa0.v, false, fb1.v,
                                                      (short)0, a01.v, false, false);
      a10.v = __builtin_amdgcn_wmma_f32_16x16x32_bf16(false, fa1.v, false, fb0.v,
                                                      (short)0, a10.v, false, false);
      a11.v = __builtin_amdgcn_wmma_f32_16x16x32_bf16(false, fa1.v, false, fb1.v,
                                                      (short)0, a11.v, false, false);
    }
  } else {
    // -------- edge path: scalar-guarded sub-tiles (small N/M launches) ----
    for (int k0 = 0; k0 < K; k0 += 32) {
      FragBF fa0, fa1, fb0, fb1;
      fa0.q[0] = *(const uint4*)(ap0);
      fa0.q[1] = *(const uint4*)(ap0 + 16);
      fb0.q[0] = *(const uint4*)(bp0);
      fb0.q[1] = *(const uint4*)(bp0 + 16);
      if (m1ok) { fa1.q[0] = *(const uint4*)(ap1); fa1.q[1] = *(const uint4*)(ap1 + 16); }
      if (n1ok) { fb1.q[0] = *(const uint4*)(bp1); fb1.q[1] = *(const uint4*)(bp1 + 16); }
      ap0 += 32; ap1 += 32; bp0 += 32; bp1 += 32;
      a00.v = __builtin_amdgcn_wmma_f32_16x16x32_bf16(false, fa0.v, false, fb0.v,
                                                      (short)0, a00.v, false, false);
      if (n1ok)
        a01.v = __builtin_amdgcn_wmma_f32_16x16x32_bf16(false, fa0.v, false, fb1.v,
                                                        (short)0, a01.v, false, false);
      if (m1ok)
        a10.v = __builtin_amdgcn_wmma_f32_16x16x32_bf16(false, fa1.v, false, fb0.v,
                                                        (short)0, a10.v, false, false);
      if (m1ok && n1ok)
        a11.v = __builtin_amdgcn_wmma_f32_16x16x32_bf16(false, fa1.v, false, fb1.v,
                                                        (short)0, a11.v, false, false);
    }
  }

  auto store_tile = [&](FragF& ac, int tm, int tn) {
    int col = tn + l15;
    bool cok = col < N;
    float bv = (bias != nullptr && cok) ? bias[col] : 0.0f;
#pragma unroll
    for (int p = 0; p < 8; ++p) {
      int row = tm + hi * 8 + p;            // C layout: vgpr p -> M=p / M=8+p
      if (cok && row < M) {
        float v = ac.f[p] * scale + bv;
        if (flags & ACT_RELU) v = fmaxf(v, 0.0f);
        long long off = (long long)row * ldc + (long long)col * cstride;
        if (flags & F_ACCUM) C[off] += v; else C[off] = v;
      }
    }
  };
  store_tile(a00, tileM, tileN);
  if (n1ok) store_tile(a01, tileM, tileN + 16);
  if (m1ok) store_tile(a10, tileM + 16, tileN);
  if (m1ok && n1ok) store_tile(a11, tileM + 16, tileN + 16);
}

// -------------------- LayerNorm: one wave per row, bf16 out -----------
__global__ __launch_bounds__(32)
void k_ln(const float* __restrict__ x, const float* __restrict__ g,
          const float* __restrict__ be, unsigned short* __restrict__ y, int C)
{
  long long row = blockIdx.x;
  x += row * C; y += row * C;
  int lane = threadIdx.x;
  float s = 0.f;
  for (int c = lane; c < C; c += 32) s += x[c];
  for (int o = 16; o > 0; o >>= 1) s += __shfl_xor(s, o, 32);
  float mean = s / C, v = 0.f;
  for (int c = lane; c < C; c += 32) { float d = x[c] - mean; v += d * d; }
  for (int o = 16; o > 0; o >>= 1) v += __shfl_xor(v, o, 32);
  float inv = rsqrtf(v / C + 1e-5f);
  for (int c = lane; c < C; c += 32) y[c] = f2bf((x[c] - mean) * inv * g[c] + be[c]);
}

// ---- softmax over keys (Lk in {64,256}); optional triangle bias ------
// bias (if set) indexed [(row*256 + k)*16 + (z&7)]  (padded 8->16 cols)
__global__ __launch_bounds__(32)
void k_softmax(const float* __restrict__ S, unsigned short* __restrict__ P, int Lk,
               long long sS, long long sP, const float* __restrict__ bias)
{
  int row = blockIdx.x, z = blockIdx.y, lane = threadIdx.x;
  const float* s = S + (long long)z * sS + (long long)row * Lk;
  unsigned short* p = P + (long long)z * sP + (long long)row * Lk;
  int nk = Lk >> 5, h = z & 7;
  float ev[8];
  float mx = -3.0e38f;
  for (int t = 0; t < nk; ++t) {
    int k = (t << 5) + lane;
    float v = s[k];
    if (bias) v += bias[((long long)row * 256 + k) * 16 + h];
    ev[t] = v; mx = fmaxf(mx, v);
  }
  for (int o = 16; o > 0; o >>= 1) mx = fmaxf(mx, __shfl_xor(mx, o, 32));
  float sum = 0.f;
  for (int t = 0; t < nk; ++t) { ev[t] = __expf(ev[t] - mx); sum += ev[t]; }
  for (int o = 16; o > 0; o >>= 1) sum += __shfl_xor(sum, o, 32);
  float inv = 1.0f / sum;
  for (int t = 0; t < nk; ++t) p[(t << 5) + lane] = f2bf(ev[t] * inv);
}

// -------------------- elementwise / repack kernels --------------------
__global__ void k_f2bf(const float* __restrict__ x, unsigned short* __restrict__ y, long long n) {
  long long t = (long long)blockIdx.x * 256 + threadIdx.x;
  if (t < n) y[t] = f2bf(x[t]);
}

// W f32 [K,N] -> Wt bf16 [Npad,K], zero padding rows
__global__ void k_wt(const float* __restrict__ W, unsigned short* __restrict__ Wt,
                     int K, int N, int Npad, long long total) {
  long long t = (long long)blockIdx.x * 256 + threadIdx.x;
  if (t >= total) return;
  int k = (int)(t % K), n = (int)(t / K);
  Wt[t] = (n < N) ? f2bf(W[(long long)k * N + n]) : (unsigned short)0;
}

// OPM weight [1024,128] (x*32+y, d) -> bf16 [(x*128+d), y=0..31]
__global__ void k_opm_w(const float* __restrict__ W, unsigned short* __restrict__ o) {
  int t = blockIdx.x * 256 + threadIdx.x;
  if (t >= 131072) return;
  int y = t & 31, n = t >> 5, x = n >> 7, d = n & 127;
  o[t] = f2bf(W[(x * 32 + y) * 128 + d]);
}

// T f32 [J][x=32][d=128] -> Tt bf16 [J][d=128][x=32]
__global__ void k_repack_T(const float* __restrict__ T, unsigned short* __restrict__ Tt,
                           long long total) {
  long long t = (long long)blockIdx.x * 256 + threadIdx.x;
  if (t >= total) return;
  long long j = t >> 12; int r = (int)(t & 4095), d = r >> 5, x = r & 31;
  Tt[t] = f2bf(T[(j << 12) + x * 128 + d]);
}

// heads: X f32 [R, c*8+h] -> bf16 [8][R][32]
__global__ void k_heads_h(const float* __restrict__ X, unsigned short* __restrict__ O,
                          int R, long long total) {
  long long t = (long long)blockIdx.x * 256 + threadIdx.x;
  if (t >= total) return;
  int c = (int)(t & 31); long long rh = t >> 5;
  int r = (int)(rh % R), h = (int)(rh / R);
  O[t] = f2bf(X[(long long)r * 256 + c * 8 + h]);
}
// heads (V transposed): -> bf16 [8][32][R]
__global__ void k_heads_h_vt(const float* __restrict__ X, unsigned short* __restrict__ O,
                             int R, long long total) {
  long long t = (long long)blockIdx.x * 256 + threadIdx.x;
  if (t >= total) return;
  int r = (int)(t % R); long long tc = t / R;
  int c = (int)(tc & 31), h = (int)(tc >> 5);
  O[t] = f2bf(X[(long long)r * 256 + c * 8 + h]);
}
// heads per-conformer: X [m*64+i, c*8+h] -> bf16 [4][8][64][32]
__global__ void k_heads_mh(const float* __restrict__ X, unsigned short* __restrict__ O) {
  int t = blockIdx.x * 256 + threadIdx.x;
  if (t >= 65536) return;
  int c = t & 31, i = (t >> 5) & 63, h = (t >> 11) & 7, m = t >> 14;
  O[t] = f2bf(X[(m * 64 + i) * 256 + c * 8 + h]);
}
// -> bf16 [4][8][32][64] (V transposed)
__global__ void k_heads_mh_vt(const float* __restrict__ X, unsigned short* __restrict__ O) {
  int t = blockIdx.x * 256 + threadIdx.x;
  if (t >= 65536) return;
  int i = t & 63, c = (t >> 6) & 31, h = (t >> 11) & 7, m = t >> 14;
  O[t] = f2bf(X[(m * 64 + i) * 256 + c * 8 + h]);
}
// triangle Q/K: X [(i*256+j), c*8+h] -> bf16 [(j*8+h)][i=256][c=32]
__global__ void k_tri_qk(const float* __restrict__ X, unsigned short* __restrict__ O,
                         long long total) {
  long long t = (long long)blockIdx.x * 256 + threadIdx.x;
  if (t >= total) return;
  int c = (int)(t & 31), i = (int)((t >> 5) & 255), h = (int)((t >> 13) & 7), j = (int)(t >> 16);
  O[t] = f2bf(X[((long long)i * 256 + j) * 256 + c * 8 + h]);
}
// triangle V: -> bf16 [(j*8+h)][c=32][k=256]
__global__ void k_tri_v(const float* __restrict__ X, unsigned short* __restrict__ O,
                        long long total) {
  long long t = (long long)blockIdx.x * 256 + threadIdx.x;
  if (t >= total) return;
  int k = (int)(t & 255), c = (int)((t >> 8) & 31), h = (int)((t >> 13) & 7), j = (int)(t >> 16);
  O[t] = f2bf(X[((long long)k * 256 + j) * 256 + c * 8 + h]);
}

__global__ void k_mean4(const float* __restrict__ X, float* __restrict__ O) {
  int t = blockIdx.x * 256 + threadIdx.x;
  if (t >= 2048) return;
  O[t] = 0.25f * (X[t] + X[t + 2048] + X[t + 4096] + X[t + 6144]);
}

// row-attn: pre-fill scores[m*8+h][i][j] with region pair-bias (fac baked in)
__global__ void k_ra_bias_fill(float* __restrict__ S, const float* __restrict__ rr,
                               const float* __restrict__ rl, const float* __restrict__ lr,
                               const float* __restrict__ llb) {
  long long t = (long long)blockIdx.x * 256 + threadIdx.x;
  if (t >= 2097152) return;
  int z = (int)(t >> 16), h = z & 7;
  int rest = (int)(t & 65535), i = rest >> 8, j = rest & 255;
  const float* src = (i < 192) ? (j < 192 ? rr : rl) : (j < 192 ? lr : llb);
  S[t] = src[(i * 256 + j) * 16 + h];
}

__global__ void k_row_combine_rec(const float* __restrict__ oreck,
                                  const float* __restrict__ olig, float* __restrict__ out) {
  int t = blockIdx.x * 256 + threadIdx.x;
  if (t >= 192 * 256) return;
  int i = t >> 8, ch = t & 255, c = ch >> 3, h = ch & 7;
  float v = oreck[h * 8192 + i * 32 + c];
  float s = 0.f;
#pragma unroll
  for (int m = 0; m < 4; ++m) s += olig[(m * 8 + h) * 8192 + i * 32 + c];
  out[t] = v + 0.25f * s;
}
__global__ void k_row_combine_lig(const float* __restrict__ oreck,
                                  const float* __restrict__ olig, float* __restrict__ out) {
  int t = blockIdx.x * 256 + threadIdx.x;
  if (t >= 256 * 256) return;
  int row = t >> 8, ch = t & 255, c = ch >> 3, h = ch & 7;
  int m = row >> 6, il = row & 63;
  int idx = ((m * 8 + h) * 256 + 192 + il) * 32 + c;
  out[t] = olig[idx] + oreck[idx];
}
__global__ void k_lc_combine(const float* __restrict__ o2, float* __restrict__ out) {
  int t = blockIdx.x * 256 + threadIdx.x;
  if (t >= 256 * 256) return;
  int row = t >> 8, ch = t & 255, c = ch >> 3, h = ch & 7;
  int m = row >> 6, i = row & 63;
  out[t] = o2[((m * 8 + h) * 64 + i) * 32 + c];
}
// triangle: O[z][i][c] * sigmoid(gate) -> attn[(i*256+j)*256 + c*8+h]
__global__ void k_tri_scatter(const float* __restrict__ O, const float* __restrict__ gate,
                              float* __restrict__ attn, int jc0, long long total) {
  long long t = (long long)blockIdx.x * 256 + threadIdx.x;
  if (t >= total) return;
  int c = (int)(t & 31), i = (int)((t >> 5) & 255), z = (int)(t >> 13);
  int j = jc0 + (z >> 3), h = z & 7;
  long long d = ((long long)i * 256 + j) * 256 + c * 8 + h;
  float g = 1.0f / (1.0f + __expf(-gate[d]));
  attn[d] = O[t] * g;
}

// =====================================================================
extern "C" void kernel_launch(void* const* d_in, const int* in_sizes, int n_in,
                              void* d_out, int out_size, void* d_ws, size_t ws_size,
                              hipStream_t stream)
{
  (void)in_sizes; (void)n_in; (void)out_size; (void)ws_size;
  const float* in_rec  = (const float*)d_in[0];
  const float* in_lig  = (const float*)d_in[1];
  const float* in_pair = (const float*)d_in[2];
  auto P = [&](int i) { return (const float*)d_in[i]; };

  float* rec_w  = (float*)d_out;            // [192,256]
  float* lig_w  = rec_w + 192 * 256;        // [4*64,256]
  float* pair_w = lig_w + 256 * 256;        // [256,256,128]

  char* cur = (char*)d_ws;
  auto allocB = [&](size_t b) { void* r = cur; cur += (b + 255) & ~(size_t)255; return r; };
  auto aF = [&](size_t n) { return (float*)allocB(n * 4); };
  auto aH = [&](size_t n) { return (unsigned short*)allocB(n * 2); };

  // transposed bf16 weights
  unsigned short *W_RAQR = aH(65536), *W_RAKR = aH(65536), *W_RAVR = aH(65536);
  unsigned short *W_RAQL = aH(65536), *W_RAKL = aH(65536), *W_RAVL = aH(65536);
  unsigned short *W_RRP = aH(2048), *W_RLP = aH(2048), *W_LRP = aH(2048), *W_LLP = aH(2048);
  unsigned short *W_FINR = aH(65536), *W_FINL = aH(65536);
  unsigned short *W_LCQ = aH(65536), *W_LCK = aH(65536), *W_LCV = aH(65536), *W_LCF = aH(65536);
  unsigned short *W_OR1 = aH(8192), *W_OR2 = aH(8192), *W_OL1 = aH(8192), *W_OL2 = aH(8192);
  unsigned short *W2RR = aH(131072), *W2RL = aH(131072), *W2LR = aH(131072), *W2LL = aH(131072);
  unsigned short *W_TR1 = aH(262144), *W_TR2 = aH(262144);
  unsigned short *W_TL1 = aH(262144), *W_TL2 = aH(262144);
  unsigned short *W_TP1 = aH(65536),  *W_TP2 = aH(65536);
  unsigned short *W_TEQ = aH(32768), *W_TEK = aH(32768), *W_TEV = aH(32768), *W_TEG = aH(32768);
  unsigned short *W_TEB = aH(2048),  *W_TEO = aH(32768);
  // activations / scratch
  unsigned short *PAIRBF = aH(8388608);
  float *RABRR = aF(1048576), *RABRL = aF(1048576), *RABLR = aF(1048576), *RABLL = aF(1048576);
  unsigned short *RECN = aH(49152), *LIGN = aH(65536);
  float *TMPS = aF(65536);
  unsigned short *RQH = aH(49152), *RKH = aH(49152), *RVT = aH(49152);
  unsigned short *LQH = aH(65536), *LKH = aH(65536), *LVT = aH(65536);
  float *RASC = aF(2097152); unsigned short *RAPR = aH(2097152);
  float *ORECK = aF(262144), *OLIGO = aF(262144);
  float *RECATT = aF(49152); unsigned short *RECATTB = aH(49152);
  float *LIGATT = aF(65536); unsigned short *LIGATTB = aH(65536);
  unsigned short *LCN = aH(65536), *QH2 = aH(65536), *KH2 = aH(65536), *VTH2 = aH(65536);
  float *SC2 = aF(131072); unsigned short *PR2 = aH(131072); float *O2 = aF(65536);
  float *LCATT = aF(65536); unsigned short *LCATTB = aH(65536);
  unsigned short *XNR = aH(49152); float *T1R = aF(196608); unsigned short *T1RB = aH(196608);
  unsigned short *XNL = aH(65536); float *T1L = aF(262144); unsigned short *T1LB = aH(262144);
  unsigned short *RECBF = aH(49152), *LIGBF = aH(65536);
  float *RI = aF(6144), *RJ = aF(6144), *LI = aF(8192), *LJ = aF(8192);
  float *LIM = aF(2048), *LJM = aF(2048);
  unsigned short *RIB = aH(6144), *RJB = aH(6144), *LIB = aH(8192), *LJB = aH(8192);
  unsigned short *LIMB = aH(2048), *LJMB = aH(2048);
  float *TBUF = aF(1048576); unsigned short *TT = aH(1048576);
  unsigned short *PAIRN = aH(8388608);
  float *BIG0 = aF(16777216), *BIG1 = aF(16777216);
  unsigned short *QH = aH(16777216), *KH = aH(16777216), *VTH = aH(16777216);
  float *BIASB = aF(1048576);
  float *TSC = aF(8388608); unsigned short *TPR = aH(8388608); float *OCH = aF(1048576);
  unsigned short *PTRN = aH(8388608);
  float *T1P = aF(8388608); unsigned short *T1PB = aH(8388608);

  auto ew = [](long long n) { return dim3((unsigned)((n + 255) / 256)); };
  auto gemm = [&](const unsigned short* A, const unsigned short* Bt, float* C,
                  int M, int N, int K, int lda, int ldb, long long ldc, int cs,
                  const float* bias, float scale, int flags, int batch,
                  long long sA, long long sB, long long sC) {
    dim3 g((N + 63) / 64, (M + 63) / 64, batch);
    k_gemm<<<g, 128, 0, stream>>>(A, Bt, C, M, N, K, lda, ldb, ldc, cs,
                                  bias, scale, flags, sA, sB, sC);
  };
  auto wt = [&](int idx, unsigned short* dst, int K, int N, int Npad) {
    long long tot = (long long)Npad * K;
    k_wt<<<ew(tot), 256, 0, stream>>>(P(idx), dst, K, N, Npad, tot);
  };
  auto f2b = [&](const float* x, unsigned short* y, long long n) {
    k_f2bf<<<ew(n), 256, 0, stream>>>(x, y, n);
  };

  // residual bases live in d_out
  hipMemcpyAsync(rec_w,  in_rec,  (size_t)49152 * 4,   hipMemcpyDeviceToDevice, stream);
  hipMemcpyAsync(lig_w,  in_lig,  (size_t)65536 * 4,   hipMemcpyDeviceToDevice, stream);
  hipMemcpyAsync(pair_w, in_pair, (size_t)8388608 * 4, hipMemcpyDeviceToDevice, stream);

  // ---- weight prep (transpose + bf16) ----
  wt(5, W_RAQR, 256, 256, 256); wt(6, W_RAKR, 256, 256, 256); wt(7, W_RAVR, 256, 256, 256);
  wt(8, W_RAQL, 256, 256, 256); wt(9, W_RAKL, 256, 256, 256); wt(10, W_RAVL, 256, 256, 256);
  wt(11, W_RRP, 128, 8, 16); wt(12, W_RLP, 128, 8, 16); wt(13, W_LRP, 128, 8, 16); wt(14, W_LLP, 128, 8, 16);
  wt(15, W_FINR, 256, 256, 256); wt(17, W_FINL, 256, 256, 256);
  wt(21, W_LCQ, 256, 256, 256); wt(22, W_LCK, 256, 256, 256); wt(23, W_LCV, 256, 256, 256);
  wt(24, W_LCF, 256, 256, 256);
  wt(26, W_OR1, 256, 32, 32); wt(28, W_OR2, 256, 32, 32);
  wt(30, W_OL1, 256, 32, 32); wt(32, W_OL2, 256, 32, 32);
  k_opm_w<<<ew(131072), 256, 0, stream>>>(P(34), W2RR);
  k_opm_w<<<ew(131072), 256, 0, stream>>>(P(36), W2RL);
  k_opm_w<<<ew(131072), 256, 0, stream>>>(P(38), W2LR);
  k_opm_w<<<ew(131072), 256, 0, stream>>>(P(40), W2LL);
  wt(44, W_TR1, 256, 1024, 1024); wt(46, W_TR2, 1024, 256, 256);
  wt(50, W_TL1, 256, 1024, 1024); wt(52, W_TL2, 1024, 256, 256);
  wt(56, W_TP1, 128, 512, 512);   wt(58, W_TP2, 512, 128, 128);
  wt(62, W_TEQ, 128, 256, 256); wt(63, W_TEK, 128, 256, 256); wt(64, W_TEV, 128, 256, 256);
  wt(65, W_TEG, 128, 256, 256); wt(67, W_TEB, 128, 8, 16);    wt(68, W_TEO, 256, 128, 128);

  // ================= RowAttentionWithPairBias =================
  k_ln<<<192, 32, 0, stream>>>(in_rec, P(3), P(4), RECN, 256);
  k_ln<<<256, 32, 0, stream>>>(in_lig, P(3), P(4), LIGN, 256);
  f2b(in_pair, PAIRBF, 8388608);
  gemm(PAIRBF, W_RRP, RABRR, 65536, 8, 128, 128, 128, 16, 1, nullptr, FAC, 0, 1, 0, 0, 0);
  gemm(PAIRBF, W_RLP, RABRL, 65536, 8, 128, 128, 128, 16, 1, nullptr, FAC, 0, 1, 0, 0, 0);
  gemm(PAIRBF, W_LRP, RABLR, 65536, 8, 128, 128, 128, 16, 1, nullptr, FAC, 0, 1, 0, 0, 0);
  gemm(PAIRBF, W_LLP, RABLL, 65536, 8, 128, 128, 128, 16, 1, nullptr, FAC, 0, 1, 0, 0, 0);
  gemm(RECN, W_RAQR, TMPS, 192, 256, 256, 256, 256, 256, 1, nullptr, 1.f, 0, 1, 0, 0, 0);
  k_heads_h<<<ew(49152), 256, 0, stream>>>(TMPS, RQH, 192, 49152);
  gemm(RECN, W_RAKR, TMPS, 192, 256, 256, 256, 256, 256, 1, nullptr, 1.f, 0, 1, 0, 0, 0);
  k_heads_h<<<ew(49152), 256, 0, stream>>>(TMPS, RKH, 192, 49152);
  gemm(RECN, W_RAVR, TMPS, 192, 256, 256, 256, 256, 256, 1, nullptr, 1.f, 0, 1, 0, 0, 0);
  k_heads_h_vt<<<ew(49152), 256, 0, stream>>>(TMPS, RVT, 192, 49152);
  gemm(LIGN, W_RAQL, TMPS, 256, 256, 256, 256, 256, 256, 1, nullptr, 1.f, 0, 1, 0, 0, 0);
  k_heads_mh<<<ew(65536), 256, 0, stream>>>(TMPS, LQH);
  gemm(LIGN, W_RAKL, TMPS, 256, 256, 256, 256, 256, 256, 1, nullptr, 1.f, 0, 1, 0, 0, 0);
  k_heads_mh<<<ew(65536), 256, 0, stream>>>(TMPS, LKH);
  gemm(LIGN, W_RAVL, TMPS, 256, 256, 256, 256, 256, 256, 1, nullptr, 1.f, 0, 1, 0, 0, 0);
  k_heads_mh_vt<<<ew(65536), 256, 0, stream>>>(TMPS, LVT);
  k_ra_bias_fill<<<ew(2097152), 256, 0, stream>>>(RASC, RABRR, RABRL, RABLR, RABLL);
  for (int m = 0; m < 4; ++m) {
    float* SC = RASC + (long long)m * 524288;
    gemm(RQH, RKH, SC, 192, 192, 32, 32, 32, 256, 1, nullptr, 1.f, F_ACCUM, 8, 6144, 6144, 65536);
    gemm(RQH, LKH + m * 16384, SC + 192, 192, 64, 32, 32, 32, 256, 1, nullptr, 1.f, F_ACCUM, 8, 6144, 2048, 65536);
    gemm(LQH + m * 16384, RKH, SC + 192 * 256, 64, 192, 32, 32, 32, 256, 1, nullptr, 1.f, F_ACCUM, 8, 2048, 6144, 65536);
    gemm(LQH + m * 16384, LKH + m * 16384, SC + 192 * 256 + 192, 64, 64, 32, 32, 32, 256, 1, nullptr, 1.f, F_ACCUM, 8, 2048, 2048, 65536);
  }
  k_softmax<<<dim3(256, 32), 32, 0, stream>>>(RASC, RAPR, 256, 65536, 65536, nullptr);
  for (int m = 0; m < 4; ++m) {
    gemm(RAPR + (long long)m * 524288, RVT, ORECK + (long long)m * 65536,
         256, 32, 192, 256, 192, 32, 1, nullptr, 1.f, 0, 8, 65536, 6144, 8192);
    gemm(RAPR + (long long)m * 524288 + 192, LVT + m * 16384, OLIGO + (long long)m * 65536,
         256, 32, 64, 256, 64, 32, 1, nullptr, 1.f, 0, 8, 65536, 2048, 8192);
  }
  k_row_combine_rec<<<ew(49152), 256, 0, stream>>>(ORECK, OLIGO, RECATT);
  k_row_combine_lig<<<ew(65536), 256, 0, stream>>>(ORECK, OLIGO, LIGATT);
  f2b(RECATT, RECATTB, 49152); f2b(LIGATT, LIGATTB, 65536);
  gemm(RECATTB, W_FINR, rec_w, 192, 256, 256, 256, 256, 256, 1, P(16), 1.f, F_ACCUM, 1, 0, 0, 0);
  gemm(LIGATTB, W_FINL, lig_w, 256, 256, 256, 256, 256, 256, 1, P(18), 1.f, F_ACCUM, 1, 0, 0, 0);

  // ================= LigColumnAttention =================
  k_ln<<<256, 32, 0, stream>>>(lig_w, P(19), P(20), LCN, 256);
  gemm(LCN, W_LCQ, TMPS, 256, 256, 256, 256, 256, 256, 1, nullptr, 1.f, 0, 1, 0, 0, 0);
  k_heads_mh<<<ew(65536), 256, 0, stream>>>(TMPS, QH2);
  gemm(LCN, W_LCK, TMPS, 256, 256, 256, 256, 256, 256, 1, nullptr, 1.f, 0, 1, 0, 0, 0);
  k_heads_mh<<<ew(65536), 256, 0, stream>>>(TMPS, KH2);
  gemm(LCN, W_LCV, TMPS, 256, 256, 256, 256, 256, 256, 1, nullptr, 1.f, 0, 1, 0, 0, 0);
  k_heads_mh_vt<<<ew(65536), 256, 0, stream>>>(TMPS, VTH2);
  gemm(QH2, KH2, SC2, 64, 64, 32, 32, 32, 64, 1, nullptr, FAC, 0, 32, 2048, 2048, 4096);
  k_softmax<<<dim3(64, 32), 32, 0, stream>>>(SC2, PR2, 64, 4096, 4096, nullptr);
  gemm(PR2, VTH2, O2, 64, 32, 64, 64, 64, 32, 1, nullptr, 1.f, 0, 32, 4096, 2048, 2048);
  k_lc_combine<<<ew(65536), 256, 0, stream>>>(O2, LCATT);
  f2b(LCATT, LCATTB, 65536);
  gemm(LCATTB, W_LCF, lig_w, 256, 256, 256, 256, 256, 256, 1, P(25), 1.f, F_ACCUM, 1, 0, 0, 0);

  // ================= Transitions (rec, lig) =================
  k_ln<<<192, 32, 0, stream>>>(rec_w, P(42), P(43), XNR, 256);
  gemm(XNR, W_TR1, T1R, 192, 1024, 256, 256, 256, 1024, 1, P(45), 1.f, ACT_RELU, 1, 0, 0, 0);
  f2b(T1R, T1RB, 196608);
  gemm(T1RB, W_TR2, rec_w, 192, 256, 1024, 1024, 1024, 256, 1, P(47), 1.f, F_ACCUM, 1, 0, 0, 0);
  k_ln<<<256, 32, 0, stream>>>(lig_w, P(48), P(49), XNL, 256);
  gemm(XNL, W_TL1, T1L, 256, 1024, 256, 256, 256, 1024, 1, P(51), 1.f, ACT_RELU, 1, 0, 0, 0);
  f2b(T1L, T1LB, 262144);
  gemm(T1LB, W_TL2, lig_w, 256, 256, 1024, 1024, 1024, 256, 1, P(53), 1.f, F_ACCUM, 1, 0, 0, 0);

  // ================= OuterProductMean (factorized) =================
  f2b(rec_w, RECBF, 49152); f2b(lig_w, LIGBF, 65536);
  gemm(RECBF, W_OR1, RI, 192, 32, 256, 256, 256, 32, 1, P(27), 1.f, 0, 1, 0, 0, 0);
  gemm(RECBF, W_OR2, RJ, 192, 32, 256, 256, 256, 32, 1, P(29), 1.f, 0, 1, 0, 0, 0);
  gemm(LIGBF, W_OL1, LI, 256, 32, 256, 256, 256, 32, 1, P(31), 1.f, 0, 1, 0, 0, 0);
  gemm(LIGBF, W_OL2, LJ, 256, 32, 256, 256, 256, 32, 1, P(33), 1.f, 0, 1, 0, 0, 0);
  k_mean4<<<ew(2048), 256, 0, stream>>>(LI, LIM);
  k_mean4<<<ew(2048), 256, 0, stream>>>(LJ, LJM);
  f2b(RI, RIB, 6144); f2b(RJ, RJB, 6144); f2b(LI, LIB, 8192); f2b(LJ, LJB, 8192);
  f2b(LIM, LIMB, 2048); f2b(LJM, LJMB, 2048);
  // rr:  T[j,x,d]=r_j@W ; pair[i,j,d] += sum_x r_i[i,x] T[j,x,d]
  gemm(RJB, W2RR, TBUF, 192, 4096, 32, 32, 32, 4096, 1, nullptr, 1.f, 0, 1, 0, 0, 0);
  k_repack_T<<<ew(786432), 256, 0, stream>>>(TBUF, TT, 786432);
  gemm(RIB, TT, pair_w, 192, 128, 32, 32, 32, 32768, 1, P(35), 1.f, F_ACCUM, 192, 0, 4096, 128);
  // rl
  gemm(LJMB, W2RL, TBUF, 64, 4096, 32, 32, 32, 4096, 1, nullptr, 1.f, 0, 1, 0, 0, 0);
  k_repack_T<<<ew(262144), 256, 0, stream>>>(TBUF, TT, 262144);
  gemm(RIB, TT, pair_w + 192 * 128, 192, 128, 32, 32, 32, 32768, 1, P(37), 1.f, F_ACCUM, 64, 0, 4096, 128);
  // lr
  gemm(LIMB, W2LR, TBUF, 64, 4096, 32, 32, 32, 4096, 1, nullptr, 1.f, 0, 1, 0, 0, 0);
  k_repack_T<<<ew(262144), 256, 0, stream>>>(TBUF, TT, 262144);
  gemm(RJB, TT, pair_w + (long long)192 * 32768, 192, 128, 32, 32, 32, 128, 1, P(39), 1.f, F_ACCUM, 64, 0, 4096, 32768);
  // ll (mean over m folded into 0.25 scale; bias added once on m==0)
  gemm(LJB, W2LL, TBUF, 256, 4096, 32, 32, 32, 4096, 1, nullptr, 1.f, 0, 1, 0, 0, 0);
  k_repack_T<<<ew(1048576), 256, 0, stream>>>(TBUF, TT, 1048576);
  for (int m = 0; m < 4; ++m)
    gemm(LIB + m * 2048, TT + (long long)m * 262144,
         pair_w + (long long)192 * 32768 + 192 * 128,
         64, 128, 32, 32, 32, 32768, 1, (m == 0 ? P(41) : nullptr), 0.25f, F_ACCUM,
         64, 0, 4096, 128);

  // ================= TriangleAttentionEndingNode =================
  k_ln<<<65536, 32, 0, stream>>>(pair_w, P(60), P(61), PAIRN, 128);
  gemm(PAIRN, W_TEQ, BIG0, 65536, 256, 128, 128, 128, 256, 1, nullptr, 1.f, 0, 1, 0, 0, 0);
  k_tri_qk<<<ew(16777216), 256, 0, stream>>>(BIG0, QH, 16777216);
  gemm(PAIRN, W_TEK, BIG0, 65536, 256, 128, 128, 128, 256, 1, nullptr, 1.f, 0, 1, 0, 0, 0);
  k_tri_qk<<<ew(16777216), 256, 0, stream>>>(BIG0, KH, 16777216);
  gemm(PAIRN, W_TEV, BIG0, 65536, 256, 128, 128, 128, 256, 1, nullptr, 1.f, 0, 1, 0, 0, 0);
  k_tri_v<<<ew(16777216), 256, 0, stream>>>(BIG0, VTH, 16777216);
  gemm(PAIRN, W_TEG, BIG1, 65536, 256, 128, 128, 128, 256, 1, P(66), 1.f, 0, 1, 0, 0, 0);
  gemm(PAIRN, W_TEB, BIASB, 65536, 8, 128, 128, 128, 16, 1, nullptr, 1.f, 0, 1, 0, 0, 0);
  for (int jc = 0; jc < 16; ++jc) {            // chunk 16 columns x 8 heads = 128 batches
    long long off = (long long)jc * 16 * 8 * 8192;
    gemm(QH + off, KH + off, TSC, 256, 256, 32, 32, 32, 256, 1, nullptr, FAC, 0, 128, 8192, 8192, 65536);
    k_softmax<<<dim3(256, 128), 32, 0, stream>>>(TSC, TPR, 256, 65536, 65536, BIASB);
    gemm(TPR, VTH + off, OCH, 256, 32, 256, 256, 256, 32, 1, nullptr, 1.f, 0, 128, 65536, 8192, 8192);
    k_tri_scatter<<<ew(1048576), 256, 0, stream>>>(OCH, BIG1, BIG0, jc * 16, 1048576);
  }
  unsigned short* ATTNB = QH;                  // QH free now; reuse as bf16 attn
  f2b(BIG0, ATTNB, 16777216);
  gemm(ATTNB, W_TEO, pair_w, 65536, 128, 256, 256, 256, 128, 1, P(69), 1.f, F_ACCUM, 1, 0, 0, 0);

  // ================= Pair transition (chunked rows) =================
  k_ln<<<65536, 32, 0, stream>>>(pair_w, P(54), P(55), PTRN, 128);
  for (int rc = 0; rc < 4; ++rc) {
    long long r0 = (long long)rc * 16384;
    gemm(PTRN + r0 * 128, W_TP1, T1P, 16384, 512, 128, 128, 128, 512, 1, P(57), 1.f, ACT_RELU, 1, 0, 0, 0);
    f2b(T1P, T1PB, 8388608);
    gemm(T1PB, W_TP2, pair_w + r0 * 128, 16384, 128, 512, 512, 512, 128, 1, P(59), 1.f, F_ACCUM, 1, 0, 0, 0);
  }
}
